// Qwen3NextGatedDeltaNet_41369124995206
// MI455X (gfx1250) — matrix operand order repaired and checked by reading, not compiled, not run on hardware
//
#include <hip/hip_runtime.h>
#include <cstdint>
#include <cstddef>

#define HIDC 2048
#define HKC 16
#define HVC 32
#define DKC 128
#define DVC 128
#define KCONV 4
#define EPSF 1e-6f
#define QKVZ_N 12288
#define BA_N 64
#define CONV_C 8192
#define CORE_N 4096

typedef __attribute__((ext_vector_type(16))) __bf16 v16bf;
typedef __attribute__((ext_vector_type(8)))  float  v8f;
typedef __attribute__((ext_vector_type(4)))  unsigned int u32x4;
typedef __attribute__((ext_vector_type(8)))  int          i32x8;
typedef __attribute__((ext_vector_type(4)))  int          i32x4;

__device__ __forceinline__ unsigned short f32_to_bf16(float f) {
    union { float f; uint32_t u; } c; c.f = f;
    uint32_t u = c.u;
    u += 0x7FFFu + ((u >> 16) & 1u);   // round-to-nearest-even
    return (unsigned short)(u >> 16);
}
__device__ __forceinline__ float bf16_to_f32(unsigned short h) {
    union { uint32_t u; float f; } c; c.u = ((uint32_t)h) << 16;
    return c.f;
}

// ---------------------------------------------------------------- cast f32->bf16
__global__ __launch_bounds__(256)
void cast_f32_bf16(const float* __restrict__ in, unsigned short* __restrict__ out, size_t n) {
    size_t i = (size_t)blockIdx.x * blockDim.x + threadIdx.x;
    size_t stride = (size_t)gridDim.x * blockDim.x;
    for (; i < n; i += stride) out[i] = f32_to_bf16(in[i]);
}

// ---------------------------------------------------------------- WMMA bf16 GEMM, double-buffered
// C[M,N] = A[M,K] * B[K,N], A/B bf16 row-major, C f32 or bf16.
// Block: 256 threads (8 waves). Tile 128x128, K-step 64, 2x LDS buffers.
// A tile staged by TDM (tensor_load_to_lds, descriptor-padded to 72-half stride);
// B tile staged transposed by the vector lanes (global loads issued before compute,
// dependent LDS scatter after compute, so HBM+DMA latency hides behind 16 WMMAs).
#define TM 128
#define TN 128
#define TK 64
#define LSTR 72   // 64 halfs payload + 8 halfs pad (TDM pad_interval=32dw, pad_amount=4dw)

union FragU { v16bf v; uint4 u[2]; };

__global__ __launch_bounds__(256)
void gemm_bf16_wmma(const unsigned short* __restrict__ A,
                    const unsigned short* __restrict__ B,
                    void* __restrict__ Cout,
                    int M, int N, int K, int c_is_bf16)
{
    __shared__ unsigned short As[2][TM * LSTR];   // A tiles [m][k], 72-half stride (TDM pad)
    __shared__ unsigned short Bs[2][TN * LSTR];   // B tiles transposed [n][k]

    const int tid  = threadIdx.x;
    const int lane = tid & 31;
    const int wave = tid >> 5;
    const int wr   = wave >> 2;        // 0..1
    const int wc   = wave & 3;         // 0..3
    const int bm   = blockIdx.y * TM;
    const int bn   = blockIdx.x * TN;

    v8f acc[4][2];
#pragma unroll
    for (int i = 0; i < 4; ++i)
#pragma unroll
        for (int j = 0; j < 2; ++j)
#pragma unroll
            for (int p = 0; p < 8; ++p) acc[i][j][p] = 0.0f;

    // ---- Tensor DMA descriptor (2-D bf16 tile), loop-invariant group 1.
    // data_size=1(2B), pad_enable, pad_interval=4 (32 DWORDs = one 64-half row),
    // pad_amount=3 (4 DWORDs = 8 halfs) -> LDS row stride 72 halfs.
    const uint32_t ldsA0 = (uint32_t)(size_t)(void*)&As[0][0];
    const uint32_t ldsA1 = (uint32_t)(size_t)(void*)&As[1][0];
    const uint32_t g1w0 = (1u << 16) | (1u << 20) | (4u << 22) | (3u << 25);
    i32x8 g1;
    g1[0] = (int)g1w0;
    g1[1] = (int)(((uint32_t)K & 0xFFFFu) << 16);                         // tensor_dim0 lo
    g1[2] = (int)(((uint32_t)K >> 16) | (((uint32_t)M & 0xFFFFu) << 16)); // dim0 hi | dim1 lo
    g1[3] = (int)(((uint32_t)M >> 16) | ((uint32_t)TK << 16));            // dim1 hi | tile_dim0
    g1[4] = (int)(uint32_t)TM;                                            // tile_dim1 | tile_dim2=0
    g1[5] = (int)(uint32_t)K;                                             // tensor_dim0_stride lo
    g1[6] = 0;
    g1[7] = 0;
    i32x4 gz4; gz4[0] = 0; gz4[1] = 0; gz4[2] = 0; gz4[3] = 0;
    i32x8 gz8;
#pragma unroll
    for (int p = 0; p < 8; ++p) gz8[p] = 0;

    const int n0 = (tid & 15) << 3;      // B staging column base
    const int kb = tid >> 4;             // B staging k base (0..15)
    uint4 breg[4];

    auto issueA = [&](int kt, int buf) {
        if (wave == 0) {
            uint64_t gaddr = (uint64_t)(size_t)A +
                (((uint64_t)(uint32_t)bm * (uint32_t)K + (uint32_t)kt) << 1);
            u32x4 g0;
            g0[0] = 1u;                                        // count=1
            g0[1] = buf ? ldsA1 : ldsA0;                       // lds_addr
            g0[2] = (uint32_t)gaddr;                           // global_addr lo
            g0[3] = (uint32_t)((gaddr >> 32) & 0x1FFFFFFu) | (2u << 30);  // hi | type=2
            __builtin_amdgcn_tensor_load_to_lds(g0, g1, gz4, gz4, gz8, 0);
        }
    };
    auto loadBreg = [&](int kt) {
#pragma unroll
        for (int p = 0; p < 4; ++p)
            breg[p] = *(const uint4*)(B + (size_t)(kt + kb + (p << 4)) * N + bn + n0);
    };
    auto scatterB = [&](int buf) {
#pragma unroll
        for (int p = 0; p < 4; ++p) {
            const unsigned short* hs = (const unsigned short*)&breg[p];
            int kl = kb + (p << 4);
#pragma unroll
            for (int j = 0; j < 8; ++j) Bs[buf][(n0 + j) * LSTR + kl] = hs[j];
        }
    };

    // ---- prologue: stage tile 0 into buffer 0
    issueA(0, 0);
    loadBreg(0);
    scatterB(0);
    if (wave == 0) __builtin_amdgcn_s_wait_tensorcnt(0);
    __syncthreads();

    const int nk = K / TK;
    const int hi = lane >> 4;
    const int lr = lane & 15;
    for (int t = 0; t < nk; ++t) {
        const int cur = t & 1;
        const int nxt = cur ^ 1;
        if (t + 1 < nk) {
            issueA((t + 1) * TK, nxt);     // DMA next A tile during compute
            loadBreg((t + 1) * TK);        // async global loads for next B tile
        }

        // ---- two K=32 WMMA steps over the staged 64-deep tile
        const unsigned short* as = &As[cur][0];
        const unsigned short* bs = &Bs[cur][0];
#pragma unroll
        for (int ks = 0; ks < TK; ks += 32) {
            v16bf afr[4], bfr[2];
#pragma unroll
            for (int i = 0; i < 4; ++i) {
                int row = wr * 64 + i * 16 + lr;
                FragU f;
                f.u[0] = *(const uint4*)&as[row * LSTR + ks + hi * 8];       // K=hi*8..+7
                f.u[1] = *(const uint4*)&as[row * LSTR + ks + hi * 8 + 16];  // K=16+hi*8..+7
                afr[i] = f.v;
            }
#pragma unroll
            for (int j = 0; j < 2; ++j) {
                int col = wc * 32 + j * 16 + lr;
                FragU f;
                f.u[0] = *(const uint4*)&bs[col * LSTR + ks + hi * 8];       // K=hi*8..+7
                f.u[1] = *(const uint4*)&bs[col * LSTR + ks + hi * 8 + 16];  // K=16+hi*8..+7
                bfr[j] = f.v;
            }
#pragma unroll
            for (int i = 0; i < 4; ++i)
#pragma unroll
                for (int j = 0; j < 2; ++j)
                    acc[i][j] = __builtin_amdgcn_wmma_f32_16x16x32_bf16(
                        false, afr[i], false, bfr[j], (short)0, acc[i][j], false, false);
        }

        if (t + 1 < nk) scatterB(nxt);     // LDS scatter of prefetched B after compute
        if (wave == 0) __builtin_amdgcn_s_wait_tensorcnt(0);
        __syncthreads();
    }

    // --- epilogue: C/D 16x16 layout: lane&15 -> col, VGPR p -> row p (+8 for hi lanes)
#pragma unroll
    for (int i = 0; i < 4; ++i) {
#pragma unroll
        for (int j = 0; j < 2; ++j) {
            int m16 = bm + wr * 64 + i * 16;
            int n16 = bn + wc * 32 + j * 16;
            int col = n16 + (lane & 15);
            int rb  = m16 + (lane >> 4) * 8;
            if (c_is_bf16) {
                unsigned short* C = (unsigned short*)Cout;
#pragma unroll
                for (int p = 0; p < 8; ++p) C[(size_t)(rb + p) * N + col] = f32_to_bf16(acc[i][j][p]);
            } else {
                float* C = (float*)Cout;
#pragma unroll
                for (int p = 0; p < 8; ++p) C[(size_t)(rb + p) * N + col] = acc[i][j][p];
            }
        }
    }
}

// ---------------------------------------------------------------- ba = hidden @ W_ba (fp32, N=64)
__global__ __launch_bounds__(64)
void ba_gemm(const float* __restrict__ hidden, const float* __restrict__ W_ba,
             float* __restrict__ ba)
{
    __shared__ float hs[HIDC];
    int row = blockIdx.x;
    int tid = threadIdx.x;
    const float* hrow = hidden + (size_t)row * HIDC;
    for (int k = tid; k < HIDC; k += 64) hs[k] = hrow[k];
    __syncthreads();
    float acc = 0.f;
    for (int k = 0; k < HIDC; ++k) acc += hs[k] * W_ba[(size_t)k * BA_N + tid];
    ba[(size_t)row * BA_N + tid] = acc;
}

// ---------------------------------------------------------------- conv1d(K=4)+SiLU (+L2norm for q/k)
__global__ __launch_bounds__(128)
void conv_silu_norm(const unsigned short* __restrict__ qkvz, const float* __restrict__ convw,
                    float* __restrict__ mixed, int S)
{
    __shared__ float red[128];
    int chunk = blockIdx.x;     // 0..63
    int s     = blockIdx.y;
    int b     = blockIdx.z;
    int d     = threadIdx.x;
    int m     = chunk * 128 + d;

    int col;
    if (m < 2048)       { col = (m >> 7) * 768 + (m & 127); }
    else if (m < 4096)  { int mm = m - 2048; col = (mm >> 7) * 768 + 128 + (mm & 127); }
    else                { int mm = m - 4096; int hv = mm >> 7;
                          col = (hv >> 1) * 768 + 256 + (hv & 1) * 128 + (mm & 127); }

    size_t base = (size_t)b * S * (size_t)QKVZ_N;
    float acc = 0.f;
#pragma unroll
    for (int i = 0; i < KCONV; ++i) {
        int ss = s - (KCONV - 1) + i;
        if (ss >= 0)
            acc += bf16_to_f32(qkvz[base + (size_t)ss * QKVZ_N + col]) * convw[m * KCONV + i];
    }
    float y = acc / (1.f + __expf(-acc));            // SiLU
    if (chunk < 32) {                                 // q/k heads: L2 normalize over 128 dims
        red[d] = y * y;
        __syncthreads();
        for (int t = 64; t > 0; t >>= 1) { if (d < t) red[d] += red[d + t]; __syncthreads(); }
        y *= rsqrtf(red[0] + EPSF);
    }
    mixed[((size_t)b * S + s) * CONV_C + m] = y;
}

// ---------------------------------------------------------------- g = -exp(A_log)*softplus(a+dt), beta = sigmoid(b)
__global__ __launch_bounds__(256)
void gate_beta(const float* __restrict__ ba, const float* __restrict__ A_log,
               const float* __restrict__ dt_bias, float* __restrict__ g,
               float* __restrict__ beta, size_t total)
{
    size_t i = (size_t)blockIdx.x * 256 + threadIdx.x;
    if (i >= total) return;
    int hv = (int)(i & 31);
    size_t bs = i >> 5;
    int hk = hv >> 1, j = hv & 1;
    float bv = ba[bs * BA_N + hk * 4 + j];
    float av = ba[bs * BA_N + hk * 4 + 2 + j];
    float x  = av + dt_bias[hv];
    float sp = (x > 20.f) ? x : log1pf(__expf(x));
    g[i]     = -__expf(A_log[hv]) * sp;
    beta[i]  = 1.f / (1.f + __expf(-bv));
}

// ---------------------------------------------------------------- gated delta scan (software-pipelined)
// One block per (b,hv). 256 threads; thread owns column v=tid&127, k-half tid>>7 (64 state regs).
// q/k/v/g/beta for step s+1 are prefetched into registers while step s computes.
__global__ __launch_bounds__(256)
void delta_scan(const float* __restrict__ mixed, const float* __restrict__ g,
                const float* __restrict__ beta, float* __restrict__ core, int S)
{
    __shared__ float qs[128], ks[128], vs[128], dlt[128], part[256], gb[2];
    int hv = blockIdx.x;
    int b  = blockIdx.y;
    int hk = hv >> 1;
    int tid = threadIdx.x;
    int v   = tid & 127;
    int ko  = (tid >> 7) * 64;

    float st[64];
#pragma unroll
    for (int j = 0; j < 64; ++j) st[j] = 0.f;

    size_t row0 = (size_t)b * S;
    float qn = 0.f, kn = 0.f, vn = 0.f, gn = 0.f, bn_ = 0.f;
    if (tid < 128) {
        qn = mixed[row0 * CONV_C + hk * 128 + tid];
        kn = mixed[row0 * CONV_C + 2048 + hk * 128 + tid];
        vn = mixed[row0 * CONV_C + 4096 + (size_t)hv * 128 + tid];
    } else if (tid == 128) {
        gn  = g[row0 * HVC + hv];
        bn_ = beta[row0 * HVC + hv];
    }

    for (int s = 0; s < S; ++s) {
        if (tid < 128) { qs[tid] = qn; ks[tid] = kn; vs[tid] = vn; }
        else if (tid == 128) { gb[0] = gn; gb[1] = bn_; }
        __syncthreads();
        if (s + 1 < S) {                 // prefetch next step (registers only, no LDS hazard)
            size_t rown = row0 + s + 1;
            if (tid < 128) {
                qn = mixed[rown * CONV_C + hk * 128 + tid];
                kn = mixed[rown * CONV_C + 2048 + hk * 128 + tid];
                vn = mixed[rown * CONV_C + 4096 + (size_t)hv * 128 + tid];
            } else if (tid == 128) {
                gn  = g[rown * HVC + hv];
                bn_ = beta[rown * HVC + hv];
            }
        }
        float decay = __expf(gb[0]);
        float bet   = gb[1];
        float p = 0.f;
#pragma unroll
        for (int j = 0; j < 64; ++j) { st[j] *= decay; p += st[j] * ks[ko + j]; }
        part[tid] = p;
        __syncthreads();
        if (tid < 128) {
            float kv = part[tid] + part[tid + 128];
            dlt[tid] = (vs[tid] - kv) * bet;
        }
        __syncthreads();
        float dd = dlt[v];
        float po = 0.f;
#pragma unroll
        for (int j = 0; j < 64; ++j) { st[j] += ks[ko + j] * dd; po += st[j] * qs[ko + j]; }
        part[tid] = po;
        __syncthreads();
        if (tid < 128) core[((row0 + s) * HVC + hv) * DVC + tid] = part[tid] + part[tid + 128];
        __syncthreads();
    }
}

// ---------------------------------------------------------------- RMSNorm * norm_w * SiLU(z), cast bf16
__global__ __launch_bounds__(128)
void rms_gate(const float* __restrict__ core, const unsigned short* __restrict__ qkvz,
              const float* __restrict__ norm_w, unsigned short* __restrict__ cg, int S)
{
    __shared__ float red[128];
    int hv = blockIdx.x, s = blockIdx.y, b = blockIdx.z;
    int d  = threadIdx.x;
    size_t row = (size_t)b * S + s;
    float c = core[(row * HVC + hv) * DVC + d];
    red[d] = c * c;
    __syncthreads();
    for (int t = 64; t > 0; t >>= 1) { if (d < t) red[d] += red[d + t]; __syncthreads(); }
    float rms = rsqrtf(red[0] * (1.f / DVC) + EPSF);
    int zcol  = (hv >> 1) * 768 + 512 + (hv & 1) * 128 + d;
    float z   = bf16_to_f32(qkvz[row * QKVZ_N + zcol]);
    float sz  = z / (1.f + __expf(-z));
    cg[row * CORE_N + hv * DVC + d] = f32_to_bf16(c * rms * norm_w[d] * sz);
}

// ================================================================ launcher
extern "C" void kernel_launch(void* const* d_in, const int* in_sizes, int n_in,
                              void* d_out, int out_size, void* d_ws, size_t ws_size,
                              hipStream_t stream)
{
    const float* hidden  = (const float*)d_in[0];
    const float* W_qkvz  = (const float*)d_in[1];
    const float* W_ba    = (const float*)d_in[2];
    const float* conv_w  = (const float*)d_in[3];
    const float* A_log   = (const float*)d_in[4];
    const float* dt_bias = (const float*)d_in[5];
    const float* norm_w  = (const float*)d_in[6];
    const float* W_out   = (const float*)d_in[7];

    const int B = 8, S = 4096;
    const size_t BS = (size_t)B * S;   // 32768 rows

    size_t off = 0;
    auto take = [&](size_t bytes) -> void* {
        void* p = (void*)((char*)d_ws + off);
        off += (bytes + 255) & ~(size_t)255;
        return p;
    };
    unsigned short* h_bf    = (unsigned short*)take(BS * HIDC * 2);
    unsigned short* wq_bf   = (unsigned short*)take((size_t)HIDC * QKVZ_N * 2);
    unsigned short* wo_bf   = (unsigned short*)take((size_t)CORE_N * HIDC * 2);
    unsigned short* qkvz_bf = (unsigned short*)take(BS * QKVZ_N * 2);
    float*          ba_f    = (float*)take(BS * BA_N * 4);
    float*          mixed_f = (float*)take(BS * CONV_C * 4);
    float*          g_f     = (float*)take(BS * HVC * 4);
    float*          beta_f  = (float*)take(BS * HVC * 4);
    float*          core_f  = (float*)take(BS * CORE_N * 4);
    unsigned short* cg_bf   = (unsigned short*)take(BS * CORE_N * 2);

    // 1) fp32 -> bf16 casts for WMMA operands
    cast_f32_bf16<<<4096, 256, 0, stream>>>(hidden, h_bf, BS * HIDC);
    cast_f32_bf16<<<4096, 256, 0, stream>>>(W_qkvz, wq_bf, (size_t)HIDC * QKVZ_N);
    cast_f32_bf16<<<2048, 256, 0, stream>>>(W_out, wo_bf, (size_t)CORE_N * HIDC);

    // 2) qkvz = hidden @ W_qkvz  (WMMA bf16, bf16 output)
    gemm_bf16_wmma<<<dim3(QKVZ_N / TN, (unsigned)(BS / TM)), 256, 0, stream>>>(
        h_bf, wq_bf, qkvz_bf, (int)BS, QKVZ_N, HIDC, 1);

    // 3) ba = hidden @ W_ba  (fp32)
    ba_gemm<<<(unsigned)BS, 64, 0, stream>>>(hidden, W_ba, ba_f);

    // 4) causal conv + SiLU + L2-norm(q,k)
    conv_silu_norm<<<dim3(CONV_C / 128, S, B), 128, 0, stream>>>(qkvz_bf, conv_w, mixed_f, S);

    // 5) decay gate g and beta
    gate_beta<<<(unsigned)((BS * HVC + 255) / 256), 256, 0, stream>>>(
        ba_f, A_log, dt_bias, g_f, beta_f, BS * HVC);

    // 6) gated delta-rule recurrence (256 independent scans)
    delta_scan<<<dim3(HVC, B), 256, 0, stream>>>(mixed_f, g_f, beta_f, core_f, S);

    // 7) RMSNorm + SiLU(z) gate, cast to bf16
    rms_gate<<<dim3(HVC, S, B), 128, 0, stream>>>(core_f, qkvz_bf, norm_w, cg_bf, S);

    // 8) out = core_gated @ W_out  (WMMA bf16, fp32 output)
    gemm_bf16_wmma<<<dim3(HIDC / TN, (unsigned)(BS / TM)), 256, 0, stream>>>(
        cg_bf, wo_bf, d_out, (int)BS, HIDC, CORE_N, 0);
}
